// SelfAttentionPooling_5858335392487
// MI455X (gfx1250) — matrix operand order repaired and checked by reading, not compiled, not run on hardware
//
#include <hip/hip_runtime.h>
#include <hip/hip_bf16.h>
#include <math.h>

#define B_ 128
#define N_ 512
#define D_ 128
#define K_ 256

typedef __attribute__((ext_vector_type(16))) _Float16 v16h;
typedef __attribute__((ext_vector_type(8)))  float    v8f;

// ---------------------------------------------------------------------------
// Kernel 1: support[b,n] = dot(X[b,n,:], w)   (wave per node, coalesced)
// ---------------------------------------------------------------------------
__global__ __launch_bounds__(256) void sap_support_kernel(const float* __restrict__ X,
                                                          const float* __restrict__ W,
                                                          float* __restrict__ support) {
    int wave = blockIdx.x * 8 + (threadIdx.x >> 5);   // node id, wave-per-node
    int lane = threadIdx.x & 31;
    const float* x = X + (size_t)wave * D_;
    float sum = 0.f;
    #pragma unroll
    for (int j = 0; j < D_; j += 32) sum += x[j + lane] * W[j + lane];
    #pragma unroll
    for (int off = 16; off > 0; off >>= 1) sum += __shfl_xor(sum, off, 32);
    if (lane == 0) support[wave] = sum;
}

// ---------------------------------------------------------------------------
// Kernel 2: attn[b,n] = tanh( (A @ support)[b,n] + bias )
// GEMV via v_wmma_f32_16x16x32_f16: one wave per 16-row tile of A.
// Fragment K-patterns are contiguous runs of 8 per lane half, so each
// operand is built from 4x float4 (global_load_b128) loads.
// B operand holds support[k] broadcast across all 16 columns, so every
// output column equals the GEMV result (column placement immaterial).
// ---------------------------------------------------------------------------
__global__ __launch_bounds__(128) void sap_score_kernel(const float* __restrict__ A,
                                                        const float* __restrict__ Svec,
                                                        const float* __restrict__ bias,
                                                        float* __restrict__ attn) {
    int wave = blockIdx.x * 4 + (threadIdx.x >> 5);
    int lane = threadIdx.x & 31;
    int b    = wave >> 5;          // N/16 = 32 tiles per graph
    int tile = wave & 31;
    int rl   = lane & 15;          // M = lane % 16 (A-matrix layout)
    bool hi  = lane >= 16;
    int klo  = hi ? 8 : 0;         // A: lanes<16 hold K [0,8)+[16,24); lanes>=16 +8
    int kbo  = hi ? 16 : 0;        // unused; B is loaded with klo, same K runs as A
    const float* Arow = A + ((size_t)b * N_ + (size_t)tile * 16 + rl) * N_;
    const float* s    = Svec + (size_t)b * N_;
    v8f c = {};
    #pragma unroll 2
    for (int kt = 0; kt < N_; kt += 32) {
        v16h av, bv;
        // A fragment: two contiguous 8-float runs per lane
        float4 a0 = *(const float4*)(Arow + kt + klo);
        float4 a1 = *(const float4*)(Arow + kt + klo + 4);
        float4 a2 = *(const float4*)(Arow + kt + 16 + klo);
        float4 a3 = *(const float4*)(Arow + kt + 16 + klo + 4);
        av[0]  = (_Float16)a0.x; av[1]  = (_Float16)a0.y;
        av[2]  = (_Float16)a0.z; av[3]  = (_Float16)a0.w;
        av[4]  = (_Float16)a1.x; av[5]  = (_Float16)a1.y;
        av[6]  = (_Float16)a1.z; av[7]  = (_Float16)a1.w;
        av[8]  = (_Float16)a2.x; av[9]  = (_Float16)a2.y;
        av[10] = (_Float16)a2.z; av[11] = (_Float16)a2.w;
        av[12] = (_Float16)a3.x; av[13] = (_Float16)a3.y;
        av[14] = (_Float16)a3.z; av[15] = (_Float16)a3.w;
        // B fragment: same two 8-value support runs as A (broadcast over columns)
        float4 b0 = *(const float4*)(s + kt + klo);
        float4 b1 = *(const float4*)(s + kt + klo + 4);
        float4 b2 = *(const float4*)(s + kt + 16 + klo);
        float4 b3 = *(const float4*)(s + kt + 16 + klo + 4);
        bv[0]  = (_Float16)b0.x; bv[1]  = (_Float16)b0.y;
        bv[2]  = (_Float16)b0.z; bv[3]  = (_Float16)b0.w;
        bv[4]  = (_Float16)b1.x; bv[5]  = (_Float16)b1.y;
        bv[6]  = (_Float16)b1.z; bv[7]  = (_Float16)b1.w;
        bv[8]  = (_Float16)b2.x; bv[9]  = (_Float16)b2.y;
        bv[10] = (_Float16)b2.z; bv[11] = (_Float16)b2.w;
        bv[12] = (_Float16)b3.x; bv[13] = (_Float16)b3.y;
        bv[14] = (_Float16)b3.z; bv[15] = (_Float16)b3.w;
        c = __builtin_amdgcn_wmma_f32_16x16x32_f16(false, av, false, bv,
                                                   (short)0, c, false, false);
    }
    float bia = bias[0];
    // C layout: VGPR r, lanes 0-15 -> M=r, lanes 16-31 -> M=8+r; cols identical.
    if (lane == 0 || lane == 16) {
        int base = b * N_ + tile * 16 + (hi ? 8 : 0);
        #pragma unroll
        for (int r = 0; r < 8; ++r) attn[base + r] = tanhf(c[r] + bia);
    }
}

// ---------------------------------------------------------------------------
// Kernel 3: exact stable top-K per graph (matches jax top_k tie-breaking),
// stable compaction -> idx already sorted ascending.
// ---------------------------------------------------------------------------
__global__ __launch_bounds__(512) void sap_topk_kernel(const float* __restrict__ attn,
                                                       int* __restrict__ idx) {
    __shared__ float vals[N_];
    __shared__ int   keep[N_];
    int b = blockIdx.x, i = threadIdx.x;
    float v = attn[b * N_ + i];
    vals[i] = v;
    __syncthreads();
    int rank = 0;
    for (int j = 0; j < N_; ++j) {
        float u = vals[j];
        rank += ((u > v) || (u == v && j < i)) ? 1 : 0;
    }
    int k = (rank < K_) ? 1 : 0;
    keep[i] = k;
    __syncthreads();
    if (k) {
        int pos = 0;
        for (int j = 0; j < i; ++j) pos += keep[j];
        idx[b * K_ + pos] = i;
    }
}

// ---------------------------------------------------------------------------
// Kernel 4: hidden[b*K+k,:] = X[b,idx,:] * attn[b,idx]; also graph indicator.
// Wave per kept node, float4 per lane -> fully coalesced 512B rows.
// ---------------------------------------------------------------------------
__global__ __launch_bounds__(256) void sap_hidden_kernel(const float* __restrict__ X,
                                                         const float* __restrict__ attn,
                                                         const int* __restrict__ idx,
                                                         float* __restrict__ hidden,
                                                         float* __restrict__ mgi) {
    int wave = blockIdx.x * 8 + (threadIdx.x >> 5);   // in [0, B*K)
    int lane = threadIdx.x & 31;
    int b    = wave >> 8;                             // K_ = 256
    int node = idx[wave];
    float a  = attn[b * N_ + node];
    const float4* src = (const float4*)(X + ((size_t)b * N_ + node) * D_);
    float4 x = src[lane];
    float4 o = make_float4(x.x * a, x.y * a, x.z * a, x.w * a);
    ((float4*)(hidden + (size_t)wave * D_))[lane] = o;
    if (lane == 0) mgi[wave] = (float)b;
}

// ---------------------------------------------------------------------------
// Kernel 5: gathered KxK binarized adjacency, +I, symmetric renormalization.
// One block per graph; K-row bitmask staged in LDS so A is gathered once
// (the gather hits L2: A + X = 167 MB < 192 MB global L2).
// ---------------------------------------------------------------------------
__global__ __launch_bounds__(256) void sap_subadj_kernel(const float* __restrict__ A,
                                                         const int* __restrict__ idx,
                                                         float* __restrict__ out) {
    __shared__ unsigned bits[K_][8];   // 8 KB: K x K adjacency bitmask
    __shared__ float    dinv[K_];
    __shared__ int      lidx[K_];
    int b = blockIdx.x, i = threadIdx.x;
    lidx[i] = idx[b * K_ + i];
    __syncthreads();
    int ii = lidx[i];
    const float* row = A + ((size_t)b * N_ + ii) * N_;
    float deg = 1.0f;                  // +I self loop
    for (int w = 0; w < 8; ++w) {
        unsigned word = 0;
        for (int t = 0; t < 32; ++t) {
            int jj = lidx[w * 32 + t];
            bool bit = (row[jj] != 0.0f) && (jj != ii);  // off-diagonal structure
            word |= ((unsigned)bit) << t;
            deg += bit ? 1.0f : 0.0f;
        }
        bits[i][w] = word;
    }
    dinv[i] = rsqrtf(deg);
    __syncthreads();
    float di = dinv[i];
    float* orow = out + ((size_t)b * K_ + i) * K_;
    for (int j = 0; j < K_; ++j) {
        unsigned bit = (bits[i][j >> 5] >> (j & 31)) & 1u;
        float ah = (bit | (unsigned)(i == j)) ? 1.0f : 0.0f;
        orow[j] = ah * di * dinv[j];
    }
}

// ---------------------------------------------------------------------------
extern "C" void kernel_launch(void* const* d_in, const int* in_sizes, int n_in,
                              void* d_out, int out_size, void* d_ws, size_t ws_size,
                              hipStream_t stream) {
    const float* A    = (const float*)d_in[0];   // [B,N,N]
    const float* X    = (const float*)d_in[1];   // [B,N,D]
    /* d_in[2] graph_indicator: unused (output is recomputed) */
    const float* W    = (const float*)d_in[3];   // [D,1]
    const float* bias = (const float*)d_in[4];   // [1]

    char*  ws      = (char*)d_ws;
    float* support = (float*)ws;                              // B*N floats
    float* attn    = (float*)(ws + (size_t)B_ * N_ * 4);      // B*N floats
    int*   idx     = (int*)  (ws + (size_t)B_ * N_ * 8);      // B*K ints

    float* hidden = (float*)d_out;                            // [B*K, D]
    float* mgi    = hidden + (size_t)B_ * K_ * D_;            // [B*K]
    float* madj   = mgi + (size_t)B_ * K_;                    // [B,K,K]

    sap_support_kernel<<<(B_ * N_) / 8, 256, 0, stream>>>(X, W, support);
    sap_score_kernel<<<(B_ * (N_ / 16)) / 4, 128, 0, stream>>>(A, support, bias, attn);
    sap_topk_kernel<<<B_, N_, 0, stream>>>(attn, idx);
    sap_hidden_kernel<<<(B_ * K_) / 8, 256, 0, stream>>>(X, attn, idx, hidden, mgi);
    sap_subadj_kernel<<<B_, K_, 0, stream>>>(A, idx, madj);
}
